// EGNNLayer_63608465654306
// MI455X (gfx1250) — matrix operand order repaired and checked by reading, not compiled, not run on hardware
//
#include <hip/hip_runtime.h>

#define DN      128
#define NNODES  40000
#define NEDGES  640000
#define LN_EPS  1e-5f

typedef __attribute__((ext_vector_type(16))) __bf16 v16bf;
typedef __attribute__((ext_vector_type(8)))  __bf16 v8bf;
typedef __attribute__((ext_vector_type(8)))  float  v8f;

// Fast SiLU: x * sigmoid(x) with hardware v_rcp_f32 (≈1ulp) instead of the
// IEEE divide expansion (v_div_scale chain) the plain '/' operator produces.
__device__ __forceinline__ float silu_f(float x) {
    return x * __builtin_amdgcn_rcpf(1.0f + __expf(-x));
}

__device__ __forceinline__ v16bf combine16(v8bf lo, v8bf hi) {
    v16bf a;
#pragma unroll
    for (int i = 0; i < 8; ++i) { a[i] = lo[i]; a[8 + i] = hi[i]; }
    return a;
}

// A-operand fragment (16x32 bf16, MxK). Lane L: row M=L%16 supplied by caller
// via `row` pointer; koff = (L<16 ? 0 : 8). Elements 0..7 <- K=kt*32+koff..+7,
// elements 8..15 <- K=kt*32+16+koff..+7 (matches ISA 16-bit A layout).
__device__ __forceinline__ v16bf load_a_bf16(const __bf16* row, int kt, int koff) {
    v8bf lo = *reinterpret_cast<const v8bf*>(row + kt * 32 + koff);
    v8bf hi = *reinterpret_cast<const v8bf*>(row + kt * 32 + 16 + koff);
    return combine16(lo, hi);
}

// B-operand fragment from pre-packed weights: flat layout
// ((kt*NT + nt)*32 + lane)*16 + i, 16 contiguous bf16 per lane.
__device__ __forceinline__ v16bf load_b_pk(const __bf16* pk, int NT, int kt, int nt, int lane) {
    const __bf16* base = pk + ((size_t)((kt * NT + nt) * 32 + lane)) * 16;
    const v8bf* p = reinterpret_cast<const v8bf*>(base);
    return combine16(p[0], p[1]);
}

__device__ __forceinline__ void atomicAddF(float* p, float v) {
    __hip_atomic_fetch_add(p, v, __ATOMIC_RELAXED, __HIP_MEMORY_SCOPE_AGENT);
}

// ---------------- setup kernels ----------------

__global__ void f32_to_bf16_kernel(const float* __restrict__ in, __bf16* __restrict__ out, int n) {
    int i = blockIdx.x * blockDim.x + threadIdx.x;
    if (i < n) out[i] = (__bf16)in[i];
}

// Pack W[rows=KT*32][128] (row-major f32) into per-lane B-fragment order (bf16).
__global__ void pack_weight_kernel(const float* __restrict__ W, __bf16* __restrict__ out, int total) {
    int idx = blockIdx.x * blockDim.x + threadIdx.x;
    if (idx >= total) return;
    int i    = idx & 15;
    int lane = (idx >> 4) & 31;
    int nt   = (idx >> 9) & 7;
    int kt   = idx >> 12;
    int row  = kt * 32 + (i & 7) + ((i >> 3) << 4) + ((lane >> 4) << 3);
    int c    = nt * 16 + (lane & 15);
    out[idx] = (__bf16)W[row * DN + c];
}

// ---------------- edge kernel ----------------

struct EdgeLds {
    int    sIdx[16];
    int    dIdx[16];
    float  dist2[16];
    float  rel[48];
    float  cw[16];
    __bf16 mtile[16 * DN];
};

__global__ __launch_bounds__(256)
void egnn_edge_kernel(const __bf16* __restrict__ hbf,
                      const float*  __restrict__ x,
                      const int*    __restrict__ ei,
                      const float*  __restrict__ We1,   // f32, for row 256 (dist2 column)
                      const float*  __restrict__ be1,
                      const float*  __restrict__ be2,
                      const float*  __restrict__ bx1,
                      const float*  __restrict__ Wx2,
                      const float*  __restrict__ bx2,
                      const __bf16* __restrict__ pkWe1,
                      const __bf16* __restrict__ pkWe2,
                      const __bf16* __restrict__ pkWx1,
                      float* __restrict__ agg,
                      float* __restrict__ xnum,
                      float* __restrict__ cnt)
{
    __shared__ EdgeLds lds[8];
    const int wave = threadIdx.x >> 5;
    const int lane = threadIdx.x & 31;
    EdgeLds& L = lds[wave];
    const int e0    = (blockIdx.x * 8 + wave) * 16;
    const int col   = lane & 15;
    const int hihal = lane >> 4;
    const int koff  = hihal ? 8 : 0;
    const int mrow0 = hihal ? 8 : 0;

    if (lane < 16) {
        int e = e0 + lane;
        int s = ei[e];
        int d = ei[NEDGES + e];
        L.sIdx[lane] = s;
        L.dIdx[lane] = d;
        float r0 = x[s * 3 + 0] - x[d * 3 + 0];
        float r1 = x[s * 3 + 1] - x[d * 3 + 1];
        float r2 = x[s * 3 + 2] - x[d * 3 + 2];
        L.rel[lane * 3 + 0] = r0;
        L.rel[lane * 3 + 1] = r1;
        L.rel[lane * 3 + 2] = r2;
        L.dist2[lane] = r0 * r0 + r1 * r1 + r2 * r2;
    }
    __syncthreads();

    const int rowS = L.sIdx[col];
    const int rowD = L.dIdx[col];

    v16bf aS[4], aD[4];
#pragma unroll
    for (int kt = 0; kt < 4; ++kt) {
        aS[kt] = load_a_bf16(hbf + (size_t)rowS * DN, kt, koff);
        aD[kt] = load_a_bf16(hbf + (size_t)rowD * DN, kt, koff);
    }
    float d2v[8];
#pragma unroll
    for (int j = 0; j < 8; ++j) d2v[j] = L.dist2[mrow0 + j];

    // ---- layer 1: m1 = silu([h_s, h_d, dist2] @ We1 + be1) ----
    for (int nt = 0; nt < 8; ++nt) {
        float b  = be1[nt * 16 + col];
        float w2 = We1[256 * DN + nt * 16 + col];   // dist2 row of We1
        v8f c;
#pragma unroll
        for (int j = 0; j < 8; ++j) c[j] = b + d2v[j] * w2;
#pragma unroll
        for (int kt = 0; kt < 4; ++kt) {
            v16bf bS = load_b_pk(pkWe1, 8, kt, nt, lane);
            c = __builtin_amdgcn_wmma_f32_16x16x32_bf16(false, aS[kt], false, bS, (short)0, c, false, false);
            v16bf bD = load_b_pk(pkWe1, 8, kt + 4, nt, lane);
            c = __builtin_amdgcn_wmma_f32_16x16x32_bf16(false, aD[kt], false, bD, (short)0, c, false, false);
        }
#pragma unroll
        for (int j = 0; j < 8; ++j)
            L.mtile[(mrow0 + j) * DN + nt * 16 + col] = (__bf16)silu_f(c[j]);
    }

    // ---- layer 2: m = silu(m1 @ We2 + be2); scatter-add into agg ----
    v16bf aM[4];
#pragma unroll
    for (int kt = 0; kt < 4; ++kt) aM[kt] = load_a_bf16(L.mtile + col * DN, kt, koff);

    for (int nt = 0; nt < 8; ++nt) {
        float b = be2[nt * 16 + col];
        v8f c;
#pragma unroll
        for (int j = 0; j < 8; ++j) c[j] = b;
#pragma unroll
        for (int kt = 0; kt < 4; ++kt) {
            v16bf bW = load_b_pk(pkWe2, 8, kt, nt, lane);
            c = __builtin_amdgcn_wmma_f32_16x16x32_bf16(false, aM[kt], false, bW, (short)0, c, false, false);
        }
#pragma unroll
        for (int j = 0; j < 8; ++j) {
            float m = silu_f(c[j]);
            L.mtile[(mrow0 + j) * DN + nt * 16 + col] = (__bf16)m;   // overwrite with final m
            atomicAddF(&agg[(size_t)L.dIdx[mrow0 + j] * DN + nt * 16 + col], m);
        }
    }

    // ---- phi_x: coord_w = silu(m @ Wx1 + bx1) @ Wx2 + bx2 ----
    v16bf aM2[4];
#pragma unroll
    for (int kt = 0; kt < 4; ++kt) aM2[kt] = load_a_bf16(L.mtile + col * DN, kt, koff);

    float pw[8];
#pragma unroll
    for (int j = 0; j < 8; ++j) pw[j] = 0.0f;

    for (int nt = 0; nt < 8; ++nt) {
        float b  = bx1[nt * 16 + col];
        float wx = Wx2[nt * 16 + col];
        v8f c;
#pragma unroll
        for (int j = 0; j < 8; ++j) c[j] = b;
#pragma unroll
        for (int kt = 0; kt < 4; ++kt) {
            v16bf bW = load_b_pk(pkWx1, 8, kt, nt, lane);
            c = __builtin_amdgcn_wmma_f32_16x16x32_bf16(false, aM2[kt], false, bW, (short)0, c, false, false);
        }
#pragma unroll
        for (int j = 0; j < 8; ++j) pw[j] += silu_f(c[j]) * wx;
    }
    // reduce across the 16 lanes of each half-wave (rows j / j+8)
#pragma unroll
    for (int j = 0; j < 8; ++j) {
        pw[j] += __shfl_xor(pw[j], 1, 32);
        pw[j] += __shfl_xor(pw[j], 2, 32);
        pw[j] += __shfl_xor(pw[j], 4, 32);
        pw[j] += __shfl_xor(pw[j], 8, 32);
    }
    if (col == 0) {   // lane 0 (rows 0..7) and lane 16 (rows 8..15)
#pragma unroll
        for (int j = 0; j < 8; ++j) L.cw[mrow0 + j] = pw[j] + bx2[0];
    }
    // same-wave LDS ordering makes cw visible
    if (lane < 16) {
        float w = L.cw[lane];
        int   d = L.dIdx[lane];
        atomicAddF(&xnum[d * 3 + 0], L.rel[lane * 3 + 0] * w);
        atomicAddF(&xnum[d * 3 + 1], L.rel[lane * 3 + 1] * w);
        atomicAddF(&xnum[d * 3 + 2], L.rel[lane * 3 + 2] * w);
        atomicAddF(&cnt[d], 1.0f);
    }
}

// ---------------- node kernel ----------------

struct NodeLds { __bf16 ttile[16 * DN]; };

__global__ __launch_bounds__(256)
void egnn_node_kernel(const __bf16* __restrict__ hbf,
                      const float*  __restrict__ h,
                      const float*  __restrict__ agg,
                      const float*  __restrict__ bh1,
                      const float*  __restrict__ bh2,
                      const float*  __restrict__ gamma,
                      const float*  __restrict__ beta,
                      const __bf16* __restrict__ pkWh1,
                      const __bf16* __restrict__ pkWh2,
                      float* __restrict__ hout)
{
    __shared__ NodeLds lds[8];
    const int wave = threadIdx.x >> 5;
    const int lane = threadIdx.x & 31;
    const int tile = blockIdx.x * 8 + wave;
    if (tile >= NNODES / 16) return;   // uniform per wave; no barriers used below
    NodeLds& L = lds[wave];
    const int n0    = tile * 16;
    const int col   = lane & 15;
    const int hihal = lane >> 4;
    const int koff  = hihal ? 8 : 0;
    const int mrow0 = hihal ? 8 : 0;
    const int rowA  = n0 + col;

    v16bf aH[4], aG[4];
#pragma unroll
    for (int kt = 0; kt < 4; ++kt) {
        aH[kt] = load_a_bf16(hbf + (size_t)rowA * DN, kt, koff);
        const float* gp = agg + (size_t)rowA * DN + kt * 32;
        v16bf a;
#pragma unroll
        for (int i = 0; i < 8; ++i) {
            a[i]     = (__bf16)gp[koff + i];
            a[8 + i] = (__bf16)gp[16 + koff + i];
        }
        aG[kt] = a;
    }

    // ---- t = silu([h, agg] @ Wh1 + bh1) ----
    for (int nt = 0; nt < 8; ++nt) {
        float b = bh1[nt * 16 + col];
        v8f c;
#pragma unroll
        for (int j = 0; j < 8; ++j) c[j] = b;
#pragma unroll
        for (int kt = 0; kt < 4; ++kt) {
            v16bf bW = load_b_pk(pkWh1, 8, kt, nt, lane);
            c = __builtin_amdgcn_wmma_f32_16x16x32_bf16(false, aH[kt], false, bW, (short)0, c, false, false);
            v16bf bG = load_b_pk(pkWh1, 8, kt + 4, nt, lane);
            c = __builtin_amdgcn_wmma_f32_16x16x32_bf16(false, aG[kt], false, bG, (short)0, c, false, false);
        }
#pragma unroll
        for (int j = 0; j < 8; ++j)
            L.ttile[(mrow0 + j) * DN + nt * 16 + col] = (__bf16)silu_f(c[j]);
    }

    // ---- hn = h + (t @ Wh2 + bh2); LayerNorm ----
    v16bf aT[4];
#pragma unroll
    for (int kt = 0; kt < 4; ++kt) aT[kt] = load_a_bf16(L.ttile + col * DN, kt, koff);

    v8f   hn[8];
    float s1[8], s2[8];
#pragma unroll
    for (int j = 0; j < 8; ++j) { s1[j] = 0.0f; s2[j] = 0.0f; }

    for (int nt = 0; nt < 8; ++nt) {
        float b = bh2[nt * 16 + col];
        v8f c;
#pragma unroll
        for (int j = 0; j < 8; ++j) c[j] = b;
#pragma unroll
        for (int kt = 0; kt < 4; ++kt) {
            v16bf bW = load_b_pk(pkWh2, 8, kt, nt, lane);
            c = __builtin_amdgcn_wmma_f32_16x16x32_bf16(false, aT[kt], false, bW, (short)0, c, false, false);
        }
#pragma unroll
        for (int j = 0; j < 8; ++j) {
            float v = h[(size_t)(n0 + mrow0 + j) * DN + nt * 16 + col] + c[j];
            c[j] = v;
            s1[j] += v;
            s2[j] += v * v;
        }
        hn[nt] = c;
    }
#pragma unroll
    for (int j = 0; j < 8; ++j) {
        s1[j] += __shfl_xor(s1[j], 1, 32); s1[j] += __shfl_xor(s1[j], 2, 32);
        s1[j] += __shfl_xor(s1[j], 4, 32); s1[j] += __shfl_xor(s1[j], 8, 32);
        s2[j] += __shfl_xor(s2[j], 1, 32); s2[j] += __shfl_xor(s2[j], 2, 32);
        s2[j] += __shfl_xor(s2[j], 4, 32); s2[j] += __shfl_xor(s2[j], 8, 32);
    }
    float mu[8], rs[8];
#pragma unroll
    for (int j = 0; j < 8; ++j) {
        mu[j] = s1[j] * (1.0f / DN);
        float var = s2[j] * (1.0f / DN) - mu[j] * mu[j];
        rs[j] = __builtin_amdgcn_rsqf(var + LN_EPS);   // v_rsq_f32
    }
    for (int nt = 0; nt < 8; ++nt) {
        float g  = gamma[nt * 16 + col];
        float bt = beta[nt * 16 + col];
#pragma unroll
        for (int j = 0; j < 8; ++j) {
            float o = (hn[nt][j] - mu[j]) * rs[j] * g + bt;
            hout[(size_t)(n0 + mrow0 + j) * DN + nt * 16 + col] = o;
        }
    }
}

// ---------------- coordinate update ----------------

__global__ void egnn_xupd_kernel(const float* __restrict__ x,
                                 const float* __restrict__ xnum,
                                 const float* __restrict__ cnt,
                                 float* __restrict__ xout)
{
    int n = blockIdx.x * blockDim.x + threadIdx.x;
    if (n >= NNODES) return;
    float c = fmaxf(cnt[n], 1.0f);
    float r = __builtin_amdgcn_rcpf(c);
#pragma unroll
    for (int k = 0; k < 3; ++k)
        xout[n * 3 + k] = x[n * 3 + k] + xnum[n * 3 + k] * r;
}

// ---------------- launcher ----------------

extern "C" void kernel_launch(void* const* d_in, const int* in_sizes, int n_in,
                              void* d_out, int out_size, void* d_ws, size_t ws_size,
                              hipStream_t stream)
{
    (void)in_sizes; (void)n_in; (void)out_size; (void)ws_size;
    const float* h    = (const float*)d_in[0];
    const float* x    = (const float*)d_in[1];
    const int*   ei   = (const int*)  d_in[2];
    const float* We1  = (const float*)d_in[4];
    const float* be1  = (const float*)d_in[5];
    const float* We2  = (const float*)d_in[6];
    const float* be2  = (const float*)d_in[7];
    const float* Wx1  = (const float*)d_in[8];
    const float* bx1  = (const float*)d_in[9];
    const float* Wx2  = (const float*)d_in[10];
    const float* bx2  = (const float*)d_in[11];
    const float* Wh1  = (const float*)d_in[12];
    const float* bh1  = (const float*)d_in[13];
    const float* Wh2  = (const float*)d_in[14];
    const float* bh2  = (const float*)d_in[15];
    const float* gam  = (const float*)d_in[16];
    const float* bet  = (const float*)d_in[17];

    char* ws = (char*)d_ws;
    size_t off = 0;
    __bf16* hbf   = (__bf16*)(ws + off); off += (size_t)NNODES * DN * 2;
    __bf16* pkWe1 = (__bf16*)(ws + off); off += (size_t)8 * 8 * 512 * 2;
    __bf16* pkWe2 = (__bf16*)(ws + off); off += (size_t)4 * 8 * 512 * 2;
    __bf16* pkWx1 = (__bf16*)(ws + off); off += (size_t)4 * 8 * 512 * 2;
    __bf16* pkWh1 = (__bf16*)(ws + off); off += (size_t)8 * 8 * 512 * 2;
    __bf16* pkWh2 = (__bf16*)(ws + off); off += (size_t)4 * 8 * 512 * 2;
    float*  agg   = (float*)(ws + off);  off += (size_t)NNODES * DN * 4;
    float*  xnum  = (float*)(ws + off);  off += (size_t)NNODES * 3 * 4;
    float*  cnt   = (float*)(ws + off);  off += (size_t)NNODES * 4;

    // zero accumulators (agg, xnum, cnt are contiguous)
    size_t zbytes = (size_t)NNODES * DN * 4 + (size_t)NNODES * 3 * 4 + (size_t)NNODES * 4;
    hipMemsetAsync(agg, 0, zbytes, stream);

    f32_to_bf16_kernel<<<(NNODES * DN + 255) / 256, 256, 0, stream>>>(h, hbf, NNODES * DN);
    pack_weight_kernel<<<(8 * 8 * 512 + 255) / 256, 256, 0, stream>>>(We1, pkWe1, 8 * 8 * 512); // rows 0..255
    pack_weight_kernel<<<(4 * 8 * 512 + 255) / 256, 256, 0, stream>>>(We2, pkWe2, 4 * 8 * 512);
    pack_weight_kernel<<<(4 * 8 * 512 + 255) / 256, 256, 0, stream>>>(Wx1, pkWx1, 4 * 8 * 512);
    pack_weight_kernel<<<(8 * 8 * 512 + 255) / 256, 256, 0, stream>>>(Wh1, pkWh1, 8 * 8 * 512);
    pack_weight_kernel<<<(4 * 8 * 512 + 255) / 256, 256, 0, stream>>>(Wh2, pkWh2, 4 * 8 * 512);

    egnn_edge_kernel<<<NEDGES / 128, 256, 0, stream>>>(
        hbf, x, ei, We1, be1, be2, bx1, Wx2, bx2,
        pkWe1, pkWe2, pkWx1, agg, xnum, cnt);

    float* hout = (float*)d_out;
    float* xout = hout + (size_t)NNODES * DN;
    egnn_node_kernel<<<(NNODES / 16 + 7) / 8, 256, 0, stream>>>(
        hbf, h, agg, bh1, bh2, gam, bet, pkWh1, pkWh2, hout);
    egnn_xupd_kernel<<<(NNODES + 255) / 256, 256, 0, stream>>>(x, xnum, cnt, xout);
}
